// GlobAttRNNDecoder_51642686767901
// MI455X (gfx1250) — matrix operand order repaired and checked by reading, not compiled, not run on hardware
//
#include <hip/hip_runtime.h>
#include <math.h>

// ---------------------------------------------------------------------------
// GlobAttRNN decoder for MI455X (gfx1250, wave32, WMMA bf16 16x16x32)
// B=64, T_ENC=1024, T_DEC=128, H=E=512, L=2, C=32
// ---------------------------------------------------------------------------

#define BB     64
#define TENC   1024
#define TDEC   128
#define HH     512
#define CC     32

typedef __attribute__((ext_vector_type(16))) __bf16 v16bf;
typedef __attribute__((ext_vector_type(8)))  float  v8f;

struct __align__(16) U128 { unsigned int x, y, z, w; };
union Frag { v16bf v; U128 u[2]; };

__device__ __forceinline__ unsigned short f2bf(float f) {
  unsigned int u = __float_as_uint(f);
  u += 0x7FFFu + ((u >> 16) & 1u);          // round-to-nearest-even
  return (unsigned short)(u >> 16);
}
__device__ __forceinline__ float bflo(unsigned int u) { return __uint_as_float(u << 16); }
__device__ __forceinline__ float bfhi(unsigned int u) { return __uint_as_float(u & 0xFFFF0000u); }

// A fragment (16x32 bf16, §7.12.2): lane(m,half): K = {half*8..+7} U {16+half*8..+7}
__device__ __forceinline__ Frag loadA(const unsigned short* row, int kb, int half) {
  Frag f;
  f.u[0] = *(const U128*)(row + kb + half * 8);
  f.u[1] = *(const U128*)(row + kb + 16 + half * 8);
  return f;
}
// B fragment (32x16 bf16): lane(n,half): K = {half*8..+7} U {16+half*8..+7}
__device__ __forceinline__ Frag loadB(const unsigned short* row, int kb, int half) {
  Frag f;
  f.u[0] = *(const U128*)(row + kb + half * 8);
  f.u[1] = *(const U128*)(row + kb + 16 + half * 8);
  return f;
}

// ---------------------------------------------------------------------------
// f32 -> bf16 conversion (setup)
// ---------------------------------------------------------------------------
__global__ void cvt_bf16_kernel(const float* __restrict__ src,
                                unsigned short* __restrict__ dst, int n) {
  int i = blockIdx.x * 256 + threadIdx.x;
  if (i < n) dst[i] = f2bf(src[i]);
}

// ---------------------------------------------------------------------------
// RNN cell: D = tanh(A1 @ W1^T + Hc @ W2^T + b1 + b2), all [64,512], K=512
// One wave per 16x16 tile; 8 waves/block; 16 blocks. Layer 0 gathers emb rows.
// ---------------------------------------------------------------------------
__global__ void gemm_rnn_kernel(const unsigned short* __restrict__ A1,
                                const int* __restrict__ tok, int tokStride,
                                const unsigned short* __restrict__ Hc,
                                const unsigned short* __restrict__ W1,
                                const unsigned short* __restrict__ W2,
                                const float* __restrict__ b1,
                                const float* __restrict__ b2,
                                unsigned short* __restrict__ D) {
  const int lane = threadIdx.x & 31;
  const int tile = blockIdx.x * 8 + (threadIdx.x >> 5);
  const int col_tile = tile & 31;        // 32 col tiles (H=512)
  const int row_tile = tile >> 5;        // 4 row tiles (B=64)
  const int m = lane & 15, half = lane >> 4;
  const int arow_i = row_tile * 16 + m;

  const unsigned short* a1row = tok
      ? A1 + (size_t)tok[(size_t)arow_i * tokStride] * HH
      : A1 + (size_t)arow_i * HH;
  const unsigned short* a2row = Hc + (size_t)arow_i * HH;
  const unsigned short* w1row = W1 + (size_t)(col_tile * 16 + m) * HH;
  const unsigned short* w2row = W2 + (size_t)(col_tile * 16 + m) * HH;

  v8f acc = {0.f, 0.f, 0.f, 0.f, 0.f, 0.f, 0.f, 0.f};
  for (int kb = 0; kb < HH; kb += 32) {
    Frag a1f = loadA(a1row, kb, half);
    Frag w1f = loadB(w1row, kb, half);
    acc = __builtin_amdgcn_wmma_f32_16x16x32_bf16(false, a1f.v, false, w1f.v,
                                                  (short)0, acc, false, false);
    Frag a2f = loadA(a2row, kb, half);
    Frag w2f = loadB(w2row, kb, half);
    acc = __builtin_amdgcn_wmma_f32_16x16x32_bf16(false, a2f.v, false, w2f.v,
                                                  (short)0, acc, false, false);
  }
  const int col = col_tile * 16 + (lane & 15);
  const float bias = b1[col] + b2[col];
#pragma unroll
  for (int j = 0; j < 8; ++j) {
    int r = row_tile * 16 + half * 8 + j;   // D layout: lane=N, vgpr=M (§7.12.2)
    D[(size_t)r * HH + col] = f2bf(tanhf(acc[j] + bias));
  }
}

// ---------------------------------------------------------------------------
// Query GEMM: q = A @ Wq^T + bq ; writes f32 q (for attention) and bf16 into
// catA[:, 512:1024] (the "q" half of concat([ctx, q])).
// ---------------------------------------------------------------------------
__global__ void gemm_q_kernel(const unsigned short* __restrict__ A,
                              const unsigned short* __restrict__ W,
                              const float* __restrict__ bias,
                              float* __restrict__ qf,
                              unsigned short* __restrict__ catA) {
  const int lane = threadIdx.x & 31;
  const int tile = blockIdx.x * 8 + (threadIdx.x >> 5);
  const int col_tile = tile & 31;
  const int row_tile = tile >> 5;
  const int m = lane & 15, half = lane >> 4;

  const unsigned short* arow = A + (size_t)(row_tile * 16 + m) * HH;
  const unsigned short* wrow = W + (size_t)(col_tile * 16 + m) * HH;

  v8f acc = {0.f, 0.f, 0.f, 0.f, 0.f, 0.f, 0.f, 0.f};
  for (int kb = 0; kb < HH; kb += 32) {
    Frag af = loadA(arow, kb, half);
    Frag wf = loadB(wrow, kb, half);
    acc = __builtin_amdgcn_wmma_f32_16x16x32_bf16(false, af.v, false, wf.v,
                                                  (short)0, acc, false, false);
  }
  const int col = col_tile * 16 + (lane & 15);
  const float b = bias[col];
#pragma unroll
  for (int j = 0; j < 8; ++j) {
    int r = row_tile * 16 + half * 8 + j;
    float v = acc[j] + b;
    qf[(size_t)r * HH + col] = v;
    catA[(size_t)r * (2 * HH) + HH + col] = f2bf(v);
  }
}

// ---------------------------------------------------------------------------
// Output GEMM: hcur = tanh(catA[64,1024] @ Wout^T + bout), K=1024
// ---------------------------------------------------------------------------
__global__ void gemm_out_kernel(const unsigned short* __restrict__ A,
                                const unsigned short* __restrict__ W,
                                const float* __restrict__ bias,
                                unsigned short* __restrict__ Hc) {
  const int lane = threadIdx.x & 31;
  const int tile = blockIdx.x * 8 + (threadIdx.x >> 5);
  const int col_tile = tile & 31;
  const int row_tile = tile >> 5;
  const int m = lane & 15, half = lane >> 4;

  const unsigned short* arow = A + (size_t)(row_tile * 16 + m) * (2 * HH);
  const unsigned short* wrow = W + (size_t)(col_tile * 16 + m) * (2 * HH);

  v8f acc = {0.f, 0.f, 0.f, 0.f, 0.f, 0.f, 0.f, 0.f};
  for (int kb = 0; kb < 2 * HH; kb += 32) {
    Frag af = loadA(arow, kb, half);
    Frag wf = loadB(wrow, kb, half);
    acc = __builtin_amdgcn_wmma_f32_16x16x32_bf16(false, af.v, false, wf.v,
                                                  (short)0, acc, false, false);
  }
  const int col = col_tile * 16 + (lane & 15);
  const float b = bias[col];
#pragma unroll
  for (int j = 0; j < 8; ++j) {
    int r = row_tile * 16 + half * 8 + j;
    Hc[(size_t)r * HH + col] = f2bf(tanhf(acc[j] + b));
  }
}

// ---------------------------------------------------------------------------
// Attention: per-batch block. scores = q . enc_h[t], masked softmax, context.
// enc_h stays f32 and L2-resident (128 MB < 192 MB L2).
// ---------------------------------------------------------------------------
__global__ void attn_kernel(const float* __restrict__ qf,
                            const float* __restrict__ enc,
                            const int* __restrict__ mask,
                            unsigned short* __restrict__ catA) {
  __shared__ float sc[TENC];
  __shared__ float qs[HH];
  __shared__ float red[8];
  const int b = blockIdx.x;
  const int tid = threadIdx.x;
  const int lane = tid & 31, w = tid >> 5;

  for (int i = tid; i < HH; i += 256) qs[i] = qf[(size_t)b * HH + i];
  __syncthreads();

  const float* encb = enc + (size_t)b * TENC * HH;
  // scores: one wave per t (lanes split H, coalesced 2KB per row), wave-reduce
  for (int t = w; t < TENC; t += 8) {
    const float* er = encb + (size_t)t * HH + lane * 16;
    __builtin_prefetch(er + 8 * HH, 0, 1);     // next t for this wave (L2 hit)
    float s = 0.f;
#pragma unroll
    for (int k = 0; k < 16; k += 4) {
      float4 e = *(const float4*)(er + k);
      const float* qp = qs + lane * 16 + k;
      s += qp[0] * e.x + qp[1] * e.y + qp[2] * e.z + qp[3] * e.w;
    }
#pragma unroll
    for (int off = 16; off; off >>= 1) s += __shfl_xor(s, off);
    if (lane == 0) sc[t] = mask[(size_t)b * TENC + t] ? s : -1e15f;
  }
  __syncthreads();

  // block max
  float mx = -3.0e38f;
  for (int t = tid; t < TENC; t += 256) mx = fmaxf(mx, sc[t]);
#pragma unroll
  for (int off = 16; off; off >>= 1) mx = fmaxf(mx, __shfl_xor(mx, off));
  if (lane == 0) red[w] = mx;
  __syncthreads();
  mx = red[0];
#pragma unroll
  for (int i = 1; i < 8; ++i) mx = fmaxf(mx, red[i]);
  __syncthreads();

  // exp + block sum (normalize lazily via 1/sum at the end)
  float se = 0.f;
  for (int t = tid; t < TENC; t += 256) {
    float e = __expf(sc[t] - mx);
    sc[t] = e;
    se += e;
  }
#pragma unroll
  for (int off = 16; off; off >>= 1) se += __shfl_xor(se, off);
  if (lane == 0) red[w] = se;
  __syncthreads();
  se = red[0];
#pragma unroll
  for (int i = 1; i < 8; ++i) se += red[i];
  const float inv = 1.f / se;
  __syncthreads();

  // context: each thread owns 2 h columns -> coalesced float2 per t
  const int h0 = tid * 2;
  float a0 = 0.f, a1 = 0.f;
  const float* ep = encb + h0;
  for (int t = 0; t < TENC; ++t) {
    float wgt = sc[t];
    float2 e = *(const float2*)(ep + (size_t)t * HH);
    a0 += wgt * e.x;
    a1 += wgt * e.y;
  }
  catA[(size_t)b * (2 * HH) + h0]     = f2bf(a0 * inv);
  catA[(size_t)b * (2 * HH) + h0 + 1] = f2bf(a1 * inv);
}

// ---------------------------------------------------------------------------
// Prediction: logits = out @ Wp^T + bp, log_softmax over C=32 (= one wave32).
// One wave per batch row; 8 waves/block; 8 blocks.
// ---------------------------------------------------------------------------
__global__ void pred_kernel(const unsigned short* __restrict__ Ob,
                            const float* __restrict__ Wp,
                            const float* __restrict__ bp,
                            float* __restrict__ out, int step) {
  const int lane = threadIdx.x & 31;          // class index
  const int b = blockIdx.x * 8 + (threadIdx.x >> 5);
  const unsigned short* orow = Ob + (size_t)b * HH;
  const float* wrow = Wp + (size_t)lane * HH;
  float s = bp[lane];
  for (int h = 0; h < HH; h += 8) {
    U128 o = *(const U128*)(orow + h);
    float4 w0 = *(const float4*)(wrow + h);
    float4 w1 = *(const float4*)(wrow + h + 4);
    s += bflo(o.x) * w0.x + bfhi(o.x) * w0.y + bflo(o.y) * w0.z + bfhi(o.y) * w0.w;
    s += bflo(o.z) * w1.x + bfhi(o.z) * w1.y + bflo(o.w) * w1.z + bfhi(o.w) * w1.w;
  }
  float mx = s;
#pragma unroll
  for (int off = 16; off; off >>= 1) mx = fmaxf(mx, __shfl_xor(mx, off));
  float e = __expf(s - mx);
  float se = e;
#pragma unroll
  for (int off = 16; off; off >>= 1) se += __shfl_xor(se, off);
  out[((size_t)b * TDEC + step) * CC + lane] = (s - mx) - __logf(se);
}

// ---------------------------------------------------------------------------
// Host launcher
// ---------------------------------------------------------------------------
extern "C" void kernel_launch(void* const* d_in, const int* in_sizes, int n_in,
                              void* d_out, int out_size, void* d_ws, size_t ws_size,
                              hipStream_t stream) {
  (void)in_sizes; (void)n_in; (void)out_size; (void)ws_size;

  const float* h      = (const float*)d_in[0];   // [1,B,H]
  const float* enc    = (const float*)d_in[1];   // [B,T_ENC,H]
  const int*   mask   = (const int*)d_in[2];     // [B,T_ENC]
  const int*   target = (const int*)d_in[3];     // [B,T_DEC]
  const float* emb    = (const float*)d_in[4];   // [H,E]
  const float* W_ih   = (const float*)d_in[5];   // [L,H,E]
  const float* W_hh   = (const float*)d_in[6];   // [L,H,H]
  const float* b_ih   = (const float*)d_in[7];   // [L,H]
  const float* b_hh   = (const float*)d_in[8];   // [L,H]
  const float* Wq     = (const float*)d_in[9];   // [L,H,H]
  const float* bq     = (const float*)d_in[10];  // [L,H]
  const float* Wout   = (const float*)d_in[11];  // [L,H,2H]
  const float* bout   = (const float*)d_in[12];  // [L,H]
  const float* Wp     = (const float*)d_in[13];  // [C,H]
  const float* bp     = (const float*)d_in[14];  // [C]
  float* out = (float*)d_out;                    // [B,T_DEC,C]

  // workspace carve-up (bf16 copies of weights + small state, ~6 MB)
  char* ws = (char*)d_ws;
  size_t off = 0;
  auto alloc = [&](size_t bytes) {
    void* p = ws + off;
    off = (off + bytes + 255) & ~(size_t)255;
    return p;
  };
  unsigned short* emb_bf  = (unsigned short*)alloc((size_t)HH * HH * 2);
  unsigned short* Wih_bf  = (unsigned short*)alloc((size_t)2 * HH * HH * 2);
  unsigned short* Whh_bf  = (unsigned short*)alloc((size_t)2 * HH * HH * 2);
  unsigned short* Wq_bf   = (unsigned short*)alloc((size_t)2 * HH * HH * 2);
  unsigned short* Wout_bf = (unsigned short*)alloc((size_t)2 * HH * 2 * HH * 2);
  unsigned short* hcur_bf = (unsigned short*)alloc((size_t)BB * HH * 2);
  unsigned short* out_a   = (unsigned short*)alloc((size_t)BB * HH * 2);
  unsigned short* out_b   = (unsigned short*)alloc((size_t)BB * HH * 2);
  float*          qf      = (float*)alloc((size_t)BB * HH * 4);
  unsigned short* catA    = (unsigned short*)alloc((size_t)BB * 2 * HH * 2);

  auto cvt = [&](const float* s, unsigned short* d, int n) {
    cvt_bf16_kernel<<<(n + 255) / 256, 256, 0, stream>>>(s, d, n);
  };
  cvt(emb,  emb_bf,  HH * HH);
  cvt(W_ih, Wih_bf,  2 * HH * HH);
  cvt(W_hh, Whh_bf,  2 * HH * HH);
  cvt(Wq,   Wq_bf,   2 * HH * HH);
  cvt(Wout, Wout_bf, 2 * HH * 2 * HH);
  cvt(h,    hcur_bf, BB * HH);

  for (int s = 0; s < TDEC; ++s) {
    for (int l = 0; l < 2; ++l) {
      const unsigned short* A1  = (l == 0) ? emb_bf : out_a;
      const int* tok            = (l == 0) ? (target + s) : nullptr;
      unsigned short* hnew      = (l == 0) ? out_a : out_b;
      gemm_rnn_kernel<<<16, 256, 0, stream>>>(
          A1, tok, TDEC, hcur_bf,
          Wih_bf + (size_t)l * HH * HH, Whh_bf + (size_t)l * HH * HH,
          b_ih + l * HH, b_hh + l * HH, hnew);
      gemm_q_kernel<<<16, 256, 0, stream>>>(
          hnew, Wq_bf + (size_t)l * HH * HH, bq + l * HH, qf, catA);
      attn_kernel<<<BB, 256, 0, stream>>>(qf, enc, mask, catA);
      gemm_out_kernel<<<16, 256, 0, stream>>>(
          catA, Wout_bf + (size_t)l * HH * 2 * HH, bout + l * HH, hcur_bf);
    }
    pred_kernel<<<8, 256, 0, stream>>>(out_b, Wp, bp, out, s);
  }
}
